// Attention_14465449853550
// MI455X (gfx1250) — matrix operand order repaired and checked by reading, not compiled, not run on hardware
//
#include <hip/hip_runtime.h>

// ---- sizes from reference ----
#define BB 4
#define SS 2048
#define NXX 1024
#define HH 16
#define DHH 64
#define N3 3072

typedef __attribute__((ext_vector_type(16))) __bf16 v16bf;
typedef __attribute__((ext_vector_type(8)))  __bf16 v8bf;
typedef __attribute__((ext_vector_type(8)))  float  v8f;
typedef __attribute__((ext_vector_type(4)))  unsigned int v4u;
typedef __attribute__((ext_vector_type(8)))  int v8i;
typedef __attribute__((ext_vector_type(4)))  int v4i;

// LDS tile: 64 weight rows x 128 K-elems, row stride padded to 136 elems
// (TDM pad: +4 DWORDs per 64-DWORD row) -> conflict-free b128 reads.
#define LDS_STRIDE 136
#define LDS_TILE_ELEMS (64 * LDS_STRIDE)

__device__ __forceinline__ v8f wmma_bf16(v16bf a, v16bf b, v8f c) {
  return __builtin_amdgcn_wmma_f32_16x16x32_bf16(
      false, a, false, b, (short)0, c, false, false);
}

// A-fragment (16x32 bf16): lane m = lane&15; elems 0..15 -> K = hi*16+i.
// Two contiguous 16B loads per lane.
__device__ __forceinline__ v16bf load_a_frag(const __bf16* row, int k, int hi) {
  v8bf lo = *(const v8bf*)(row + k + hi * 16);
  v8bf hh = *(const v8bf*)(row + k + hi * 16 + 8);
  return __builtin_shufflevector(lo, hh, 0,1,2,3,4,5,6,7,8,9,10,11,12,13,14,15);
}

// ------------------------------------------------------------------
// TDM: load one 64-row x 128-col bf16 tile of wT[n0.., kc..] into LDS.
// D# per CDNA5 ISA ch.8: group0 = {count/lds_addr/global_addr/type},
// group1 = {flags, dims, tile dims, strides}. Pad 4 DWORDs per row.
// This toolchain exposes the 6-arg (clang-23) form of the builtin:
//   (uint32x4 g0, int32x8 g1, int32x4 g2, int32x4 g3, int32x8, i32 cpol)
__device__ __forceinline__ void tdm_load_tile(unsigned lds_off,
                                              const __bf16* gptr,
                                              int tensor_rows) {
  unsigned long long ga = (unsigned long long)(size_t)gptr;
  v4u g0;
  g0[0] = 1u;                                      // count=1, user mode
  g0[1] = lds_off;                                 // lds_addr (bytes)
  g0[2] = (unsigned)(ga & 0xffffffffu);            // global_addr[31:0]
  g0[3] = (unsigned)((ga >> 32) & 0x01ffffffu)     // global_addr[56:32]
          | (2u << 30);                            // type=2 ("image")
  v8i g1;
  g1[0] = (1 << 16)        // data_size=1 (2 bytes)
        | (1 << 20)        // pad_enable
        | (5 << 22)        // pad_interval: 64 DWORDs (one 128-elem row)
        | (3 << 25);       // pad_amount: 4 DWORDs (8 elems)
  g1[1] = (int)(1024u << 16);                 // tensor_dim0[15:0] = 1024 (K)
  g1[2] = (int)((unsigned)tensor_rows << 16); // dim0[31:16]=0 | tensor_dim1[15:0]
  g1[3] = (int)(128u << 16);                  // dim1[31:16]=0 | tile_dim0 = 128
  g1[4] = 64;                                 // tile_dim1 = 64 | tile_dim2 = 0
  g1[5] = 1024;                               // tensor_dim0_stride[31:0]
  g1[6] = 0;
  g1[7] = 0;
  v4i z4 = {0, 0, 0, 0};
  v8i z8 = {0, 0, 0, 0, 0, 0, 0, 0};
  __builtin_amdgcn_tensor_load_to_lds(g0, g1, z4, z4, z8, 0);
  __builtin_amdgcn_s_wait_tensorcnt(0);
}

// ------------------------------------------------------------------
__global__ __launch_bounds__(256) void cvt_f32_to_bf16(
    const float* __restrict__ in, __bf16* __restrict__ out, int n) {
  int i = blockIdx.x * blockDim.x + threadIdx.x;
  if (i < n) out[i] = (__bf16)in[i];
}

// w[K][N] fp32 -> wt[N][K] bf16 (B-frag rows contiguous in K)
__global__ __launch_bounds__(256) void transpose_to_bf16(
    const float* __restrict__ w, __bf16* __restrict__ wt, int K, int N) {
  int i = blockIdx.x * blockDim.x + threadIdx.x;
  if (i < K * N) {
    int k = i / N, n = i % N;
    wt[(size_t)n * K + k] = (__bf16)w[i];
  }
}

// ------------------------------------------------------------------
// Shared GEMM core: 8 waves/block, each wave does a 16x64 C tile; the
// 64-col weight supertile is staged once per 128-K chunk via the TDM.
template <int EPILOGUE>  // 0 = qkv scatter, 1 = proj f32 out
__device__ __forceinline__ void gemm_core(
    const __bf16* __restrict__ ain, const __bf16* __restrict__ wT,
    int wt_rows, const float* __restrict__ bias,
    __bf16* __restrict__ q, __bf16* __restrict__ kk, __bf16* __restrict__ vT,
    float* __restrict__ out) {
  __shared__ __bf16 lds_w[LDS_TILE_ELEMS];
  const int wid = threadIdx.x >> 5;
  const int lane = threadIdx.x & 31;
  const int ln = lane & 15, hi = lane >> 4;
  const int row0 = (blockIdx.x * 8 + wid) * 16;
  const int n0 = blockIdx.y * 64;
  const __bf16* arow = ain + (size_t)(row0 + ln) * NXX;

  v8f acc0 = {0,0,0,0,0,0,0,0}, acc1 = acc0, acc2 = acc0, acc3 = acc0;
  const unsigned lds_off = (unsigned)(size_t)&lds_w[0];

  for (int kc = 0; kc < NXX; kc += 128) {
    __syncthreads();  // previous tile fully consumed
    if (threadIdx.x < 32)  // one wave issues the DMA, EXEC-independent
      tdm_load_tile(lds_off, wT + (size_t)n0 * NXX + kc, wt_rows);
    __syncthreads();  // tile visible to all 8 waves

#pragma unroll
    for (int k2 = 0; k2 < 128; k2 += 32) {
      v16bf a = load_a_frag(arow, kc + k2, hi);
      const __bf16* lb = lds_w + k2 + hi * 16;
      acc0 = wmma_bf16(a, *(const v16bf*)(lb + (0  + ln) * LDS_STRIDE), acc0);
      acc1 = wmma_bf16(a, *(const v16bf*)(lb + (16 + ln) * LDS_STRIDE), acc1);
      acc2 = wmma_bf16(a, *(const v16bf*)(lb + (32 + ln) * LDS_STRIDE), acc2);
      acc3 = wmma_bf16(a, *(const v16bf*)(lb + (48 + ln) * LDS_STRIDE), acc3);
    }
  }

  v8f accs[4] = {acc0, acc1, acc2, acc3};
#pragma unroll
  for (int t = 0; t < 4; ++t) {
    int c = n0 + t * 16 + ln;
    float bv = bias[c];
    if (EPILOGUE == 0) {
      int sel = c >> 10, cc = c & 1023;
      int h = cc >> 6, d = cc & 63;
#pragma unroll
      for (int r = 0; r < 8; ++r) {
        int row = row0 + r + 8 * hi;
        int b_ = row >> 11, s_ = row & 2047;
        __bf16 bvv = (__bf16)(accs[t][r] + bv);
        size_t bh = (size_t)(b_ * HH + h);
        if (sel == 0)      q [(bh * SS + s_) * DHH + d] = bvv;
        else if (sel == 1) kk[(bh * SS + s_) * DHH + d] = bvv;
        else               vT[(bh * DHH + d) * SS + s_] = bvv;
      }
    } else {
#pragma unroll
      for (int r = 0; r < 8; ++r) {
        int row = row0 + r + 8 * hi;
        out[(size_t)row * NXX + c] = accs[t][r] + bv;
      }
    }
  }
}

__global__ __launch_bounds__(256) void qkv_gemm(
    const __bf16* __restrict__ xb, const __bf16* __restrict__ watT,
    const float* __restrict__ b_attn,
    __bf16* __restrict__ q, __bf16* __restrict__ kk, __bf16* __restrict__ vT) {
  gemm_core<0>(xb, watT, N3, b_attn, q, kk, vT, nullptr);
}

__global__ __launch_bounds__(256) void proj_gemm(
    const __bf16* __restrict__ ab, const __bf16* __restrict__ wpT,
    const float* __restrict__ bias, float* __restrict__ out) {
  gemm_core<1>(ab, wpT, NXX, bias, nullptr, nullptr, nullptr, out);
}

// ------------------------------------------------------------------
// Flash attention: one wave = 16 query rows of one (b,h); 32-key blocks.
// Divergent causal trip counts per wave -> no block barriers here.
__global__ __launch_bounds__(128) void attn_kernel(
    const __bf16* __restrict__ q, const __bf16* __restrict__ kk,
    const __bf16* __restrict__ vT, __bf16* __restrict__ ao) {
  __shared__ float p_lds[4][16][32];
  const int wid = threadIdx.x >> 5;
  const int lane = threadIdx.x & 31;
  const int ln = lane & 15, hi = lane >> 4;
  const int bh = blockIdx.y;
  const int qbase = blockIdx.x * 64 + wid * 16;

  const __bf16* qp = q  + (size_t)bh * SS * DHH;
  const __bf16* kp = kk + (size_t)bh * SS * DHH;
  const __bf16* vp = vT + (size_t)bh * DHH * SS;

  const __bf16* qrow = qp + (size_t)(qbase + ln) * DHH;
  v16bf qA0 = load_a_frag(qrow, 0, hi);
  v16bf qA1 = load_a_frag(qrow, 32, hi);

  v8f zero = {0,0,0,0,0,0,0,0};
  v8f acc[4] = {zero, zero, zero, zero};
  float rowm[8], rowl[8];
#pragma unroll
  for (int r = 0; r < 8; ++r) { rowm[r] = -1e30f; rowl[r] = 0.0f; }

  const int nkt = ((qbase + 15) >> 5) + 1;
  for (int kt = 0; kt < nkt; ++kt) {
    const int k0 = kt * 32;
    v8f s[2];
#pragma unroll
    for (int sub = 0; sub < 2; ++sub) {
      const int ksub = k0 + sub * 16;
      const __bf16* krow = kp + (size_t)(ksub + ln) * DHH;
      v16bf kB0 = *(const v16bf*)(krow + hi * 16);
      v16bf kB1 = *(const v16bf*)(krow + 32 + hi * 16);
      v8f t = zero;
      t = wmma_bf16(qA0, kB0, t);
      t = wmma_bf16(qA1, kB1, t);
      s[sub] = t;
    }

    float newm[8];
#pragma unroll
    for (int r = 0; r < 8; ++r) {
      int qr = qbase + r + 8 * hi;
      float a0 = s[0][r] * 0.125f;
      float a1 = s[1][r] * 0.125f;
      if (k0 + ln > qr)      a0 = -10000.0f;
      if (k0 + 16 + ln > qr) a1 = -10000.0f;
      s[0][r] = a0; s[1][r] = a1;
      float mx = fmaxf(a0, a1);
      mx = fmaxf(mx, __shfl_xor(mx, 1, 32));
      mx = fmaxf(mx, __shfl_xor(mx, 2, 32));
      mx = fmaxf(mx, __shfl_xor(mx, 4, 32));
      mx = fmaxf(mx, __shfl_xor(mx, 8, 32));
      newm[r] = fmaxf(rowm[r], mx);
    }
#pragma unroll
    for (int r = 0; r < 8; ++r) {
      float corr = __expf(rowm[r] - newm[r]);
      rowm[r] = newm[r];
      float p0 = __expf(s[0][r] - newm[r]);
      float p1 = __expf(s[1][r] - newm[r]);
      float ps = p0 + p1;
      ps += __shfl_xor(ps, 1, 32);
      ps += __shfl_xor(ps, 2, 32);
      ps += __shfl_xor(ps, 4, 32);
      ps += __shfl_xor(ps, 8, 32);
      rowl[r] = rowl[r] * corr + ps;
      acc[0][r] *= corr; acc[1][r] *= corr; acc[2][r] *= corr; acc[3][r] *= corr;
      int m = r + 8 * hi;
      p_lds[wid][m][ln] = p0;
      p_lds[wid][m][16 + ln] = p1;
    }
    asm volatile("s_wait_dscnt 0" ::: "memory");

    const float* prow = &p_lds[wid][ln][0];
    v16bf pA;
#pragma unroll
    for (int i = 0; i < 8; ++i) pA[i] = (__bf16)prow[hi * 16 + i];
#pragma unroll
    for (int i = 0; i < 8; ++i) pA[8 + i] = (__bf16)prow[hi * 16 + 8 + i];

#pragma unroll
    for (int dt = 0; dt < 4; ++dt) {
      const __bf16* vrow = vp + (size_t)(dt * 16 + ln) * SS + k0 + hi * 16;
      acc[dt] = wmma_bf16(pA, *(const v16bf*)vrow, acc[dt]);
    }
  }

  int b_ = bh >> 4, h = bh & 15;
#pragma unroll
  for (int dt = 0; dt < 4; ++dt) {
#pragma unroll
    for (int r = 0; r < 8; ++r) {
      int qr = qbase + r + 8 * hi;
      float o = acc[dt][r] / rowl[r];
      ao[((size_t)(b_ * SS + qr)) * NXX + h * DHH + dt * 16 + ln] = (__bf16)o;
    }
  }
}

// ------------------------------------------------------------------
extern "C" void kernel_launch(void* const* d_in, const int* in_sizes, int n_in,
                              void* d_out, int out_size, void* d_ws, size_t ws_size,
                              hipStream_t stream) {
  const float* x      = (const float*)d_in[0];
  const float* w_attn = (const float*)d_in[1];
  const float* b_attn = (const float*)d_in[2];
  const float* w_proj = (const float*)d_in[3];
  const float* b_proj = (const float*)d_in[4];
  float* out = (float*)d_out;

  char* ws = (char*)d_ws;
  size_t off = 0;
  auto carve = [&](size_t bytes) -> void* {
    void* p = ws + off;
    off += (bytes + 255) & ~(size_t)255;
    return p;
  };
  __bf16* xb   = (__bf16*)carve((size_t)BB * SS * NXX * 2);
  __bf16* watT = (__bf16*)carve((size_t)N3 * NXX * 2);
  __bf16* wpT  = (__bf16*)carve((size_t)NXX * NXX * 2);
  __bf16* qb   = (__bf16*)carve((size_t)BB * HH * SS * DHH * 2);
  __bf16* kb   = (__bf16*)carve((size_t)BB * HH * SS * DHH * 2);
  __bf16* vTb  = (__bf16*)carve((size_t)BB * HH * DHH * SS * 2);
  __bf16* ab   = (__bf16*)carve((size_t)BB * SS * NXX * 2);

  const int nx = BB * SS * NXX;
  cvt_f32_to_bf16<<<dim3((nx + 255) / 256), dim3(256), 0, stream>>>(x, xb, nx);
  transpose_to_bf16<<<dim3((NXX * N3 + 255) / 256), dim3(256), 0, stream>>>(
      w_attn, watT, NXX, N3);
  transpose_to_bf16<<<dim3((NXX * NXX + 255) / 256), dim3(256), 0, stream>>>(
      w_proj, wpT, NXX, NXX);

  qkv_gemm<<<dim3(64, 48), dim3(256), 0, stream>>>(xb, watT, b_attn, qb, kb, vTb);
  attn_kernel<<<dim3(SS / 64, BB * HH), dim3(128), 0, stream>>>(qb, kb, vTb, ab);
  proj_gemm<<<dim3(64, 16), dim3(256), 0, stream>>>(ab, wpT, b_proj, out);
}
